// SpatialNetVLAD_1125281431848
// MI455X (gfx1250) — matrix operand order repaired and checked by reading, not compiled, not run on hardware
//
#include <hip/hip_runtime.h>
#include <hip/hip_bf16.h>

// Problem constants
#define NIMG 32
#define HDIM 36
#define WDIM 36
#define DDIM 512
#define KCLU 64
#define NREG 9          // 3x3 pooling regions
#define RPIX 144        // 12*12 pixels per region
#define RPAD 160        // padded to 5 chunks of 32
#define EPSF 1e-12f
#define XSTRIDE_N (HDIM*WDIM*DDIM)   // 663552
#define OUT_PER_REG (DDIM*KCLU)      // 32768

typedef __attribute__((ext_vector_type(16))) _Float16 v16h;
typedef __attribute__((ext_vector_type(8)))  _Float16 v8h;
typedef __attribute__((ext_vector_type(8)))  float    v8f;
typedef __attribute__((ext_vector_type(4)))  float    v4f;

union F16Frag { v16h v; v8h h[2]; };

// Workspace layout (bytes)
#define WT_OFF   0                                        // Wt f16 [64][512]
#define AT_OFF   (KCLU*DDIM*2)                            // At f16 [288][64][160]
#define CSQ_OFF  (AT_OFF + NIMG*NREG*KCLU*RPAD*2)         // colsq f32 [288][64]

__device__ __forceinline__ v8f wmma16(v16h a, v16h b, v8f c) {
  // D = A(16x32 f16) * B(32x16 f16) + C(16x16 f32)
  return __builtin_amdgcn_wmma_f32_16x16x32_f16(false, a, false, b, (short)0, c,
                                                false, false);
}

__device__ __forceinline__ v8h cvt8(const float* p) {
  v4f a = *(const v4f*)p;
  v4f b = *(const v4f*)(p + 4);
  v8h r;
  r[0] = (_Float16)a[0]; r[1] = (_Float16)a[1];
  r[2] = (_Float16)a[2]; r[3] = (_Float16)a[3];
  r[4] = (_Float16)b[0]; r[5] = (_Float16)b[1];
  r[6] = (_Float16)b[2]; r[7] = (_Float16)b[3];
  return r;
}

// ---------------------------------------------------------------------------
// Kernel 0: Wt[c][d] = (f16) W_conv[d][c]   (B-matrix friendly layout)
// ---------------------------------------------------------------------------
__global__ __launch_bounds__(256) void k_wt(const float* __restrict__ Wc,
                                            _Float16* __restrict__ Wt) {
  int idx = blockIdx.x * 256 + threadIdx.x;   // 128 blocks * 256 = 32768
  if (idx < KCLU * DDIM) {
    int c = idx >> 9;          // /512
    int d = idx & 511;
    Wt[idx] = (_Float16)Wc[d * KCLU + c];
  }
}

// ---------------------------------------------------------------------------
// Kernel 1: per (n,region): s = X[144,512] @ W[512,64] (WMMA), softmax rows,
// store At (transposed, padded) f16 to ws.  blockDim = 288 (9 waves).
// Wave w computes M-tile w (16 pixel rows) for all 4 N-tiles.
// ---------------------------------------------------------------------------
__global__ __launch_bounds__(288) void k_assign(const float* __restrict__ x,
                                                const _Float16* __restrict__ Wt,
                                                _Float16* __restrict__ AtAll) {
  __shared__ float sLds[RPIX * KCLU];   // 36 KB
  __shared__ int   pixOff[RPIX];

  const int nr = blockIdx.x;
  const int n = nr / NREG, region = nr % NREG;
  const int jy = region / 3, ix = region % 3;
  const int tid = threadIdx.x;

  if (tid < RPIX) {
    int y = jy * 12 + tid / 12;
    int xc = ix * 12 + tid % 12;
    pixOff[tid] = (y * WDIM + xc) * DDIM;
  }
  __syncthreads();

  const int wave = tid >> 5, lane = tid & 31;
  const int lm = lane & 15, lh = lane >> 4;
  const int kb = lh * 8;
  const int m0 = wave * 16;

  const float* xn = x + (size_t)n * XSTRIDE_N;
  const float* xrow = xn + pixOff[m0 + lm];   // this lane's pixel row

  v8f acc[4];
#pragma unroll
  for (int t = 0; t < 4; ++t)
#pragma unroll
    for (int e = 0; e < 8; ++e) acc[t][e] = 0.0f;

  for (int kk = 0; kk < DDIM; kk += 32) {
    F16Frag af;
    af.h[0] = cvt8(xrow + kk + lh * 16);        // K = kk+16*lh .. +7
    af.h[1] = cvt8(xrow + kk + lh * 16 + 8);   // K = kk+16*lh+8 .. +7
#pragma unroll
    for (int nt = 0; nt < 4; ++nt) {
      F16Frag bf;
      bf.v = *(const v16h*)(Wt + (size_t)(nt * 16 + lm) * DDIM + kk + lh * 16);
      acc[nt] = wmma16(af.v, bf.v, acc[nt]);
    }
  }

  // D-tile -> LDS:  row M = m0 + 8*lh + r ; col N = nt*16 + lm
#pragma unroll
  for (int nt = 0; nt < 4; ++nt)
#pragma unroll
    for (int r = 0; r < 8; ++r)
      sLds[(m0 + 8 * lh + r) * KCLU + nt * 16 + lm] = acc[nt][r];
  __syncthreads();

  // Softmax over 64 clusters per pixel row; store transposed f16, pad zeros.
  _Float16* At = AtAll + (size_t)nr * KCLU * RPAD;
  if (tid < RPIX) {
    const float* srow = &sLds[tid * KCLU];
    float mx = srow[0];
    for (int c = 1; c < KCLU; ++c) mx = fmaxf(mx, srow[c]);
    float sum = 0.0f;
    for (int c = 0; c < KCLU; ++c) sum += __expf(srow[c] - mx);
    float inv = 1.0f / sum;
    for (int c = 0; c < KCLU; ++c)
      At[(size_t)c * RPAD + tid] = (_Float16)(__expf(srow[c] - mx) * inv);
  } else if (tid < RPAD) {
    for (int c = 0; c < KCLU; ++c)
      At[(size_t)c * RPAD + tid] = (_Float16)0.0f;
  }
}

// ---------------------------------------------------------------------------
// Kernel 2: per (n,region): Vt[64,512] = At[64,160] @ Xp[160,512]  (WMMA),
// then v -= C[d][c]*asum[c]; write raw v to out (index d*64+c); accumulate
// per-cluster sum of squares -> ws.  blockDim = 256 (8 waves).
// Wave w: cluster-tile mt = w&3, d-tiles nt in [(w>>2)*16, +16).
// ---------------------------------------------------------------------------
__global__ __launch_bounds__(256) void k_vlad(const float* __restrict__ x,
                                              const float* __restrict__ Cmat,
                                              const _Float16* __restrict__ AtAll,
                                              float* __restrict__ out,
                                              float* __restrict__ colsqAll) {
  __shared__ int   pixOff[RPAD];
  __shared__ float asumLds[KCLU];
  __shared__ float colsq[KCLU];

  const int nr = blockIdx.x;
  const int n = nr / NREG, region = nr % NREG;
  const int jy = region / 3, ix = region % 3;
  const int tid = threadIdx.x;

  const _Float16* At = AtAll + (size_t)nr * KCLU * RPAD;

  for (int p = tid; p < RPAD; p += 256) {
    int pc = p < RPIX ? p : (RPIX - 1);   // clamp (pad rows of At are zero)
    int y = jy * 12 + pc / 12;
    int xc = ix * 12 + pc % 12;
    pixOff[p] = (y * WDIM + xc) * DDIM;
  }
  if (tid < KCLU) {
    float s = 0.0f;
    for (int p = 0; p < RPIX; ++p) s += (float)At[(size_t)tid * RPAD + p];
    asumLds[tid] = s;
    colsq[tid] = 0.0f;
  }
  __syncthreads();

  const int wave = tid >> 5, lane = tid & 31;
  const int lm = lane & 15, lh = lane >> 4;
  const int kb = lh * 8;
  const int mt = wave & 3;
  const int m0 = mt * 16;
  const int ntBase = (wave >> 2) * 16;
  const int cbase = m0 + 8 * lh;          // 8 consecutive clusters per lane

  // Cache the 5 A-fragments (cluster rows, pixel K) for this wave's M-tile.
  F16Frag af[5];
  const _Float16* arow = At + (size_t)(m0 + lm) * RPAD;
#pragma unroll
  for (int kc = 0; kc < 5; ++kc) {
    int kk = kc * 32;
    af[kc].h[0] = *(const v8h*)(arow + kk + lh * 16);
    af[kc].h[1] = *(const v8h*)(arow + kk + lh * 16 + 8);
  }

  float csq[8];
#pragma unroll
  for (int r = 0; r < 8; ++r) csq[r] = 0.0f;

  const float* xn = x + (size_t)n * XSTRIDE_N;
  float* outp = out + (size_t)nr * OUT_PER_REG;

  for (int t = 0; t < 16; ++t) {
    const int d = (ntBase + t) * 16 + lm;   // this lane's feature dim
    v8f acc;
#pragma unroll
    for (int e = 0; e < 8; ++e) acc[e] = 0.0f;

#pragma unroll
    for (int kc = 0; kc < 5; ++kc) {
      const int kk = kc * 32 + lh * 16;     // 16 consecutive pixels
      F16Frag bf;
#pragma unroll
      for (int j = 0; j < 16; ++j)
        bf.v[j] = (_Float16)xn[pixOff[kk + j] + d];
      acc = wmma16(af[kc].v, bf.v, acc);
    }

    // v[d][c] = acc - C[d][c]*asum[c]; 8 consecutive clusters -> float4 x2
    const float* Cp = Cmat + (size_t)d * KCLU + cbase;
    v4f c0 = *(const v4f*)Cp;
    v4f c1 = *(const v4f*)(Cp + 4);
    v4f lo, hi;
#pragma unroll
    for (int r = 0; r < 8; ++r) {
      float cv = (r < 4) ? c0[r] : c1[r - 4];
      float vv = acc[r] - cv * asumLds[cbase + r];
      csq[r] += vv * vv;
      if (r < 4) lo[r] = vv; else hi[r - 4] = vv;
    }
    float* op = outp + (size_t)d * KCLU + cbase;
    *(v4f*)op = lo;
    *(v4f*)(op + 4) = hi;
  }

#pragma unroll
  for (int r = 0; r < 8; ++r) atomicAdd(&colsq[cbase + r], csq[r]);
  __syncthreads();

  if (tid < KCLU) colsqAll[(size_t)nr * KCLU + tid] = colsq[tid];
}

// ---------------------------------------------------------------------------
// Kernel 3: per (n,region): intra-norm (over D) + global L2 (over 32768),
// using closed form: global_norm^2 = sum_k colsq[k]/(colsq[k]+eps).
// In-place rescale of d_out.
// ---------------------------------------------------------------------------
__global__ __launch_bounds__(256) void k_norm(float* __restrict__ out,
                                              const float* __restrict__ colsqAll) {
  __shared__ float colinv[KCLU];
  __shared__ float term[KCLU];
  __shared__ float gscale;

  const int nr = blockIdx.x;
  const int tid = threadIdx.x;
  const float* csq = colsqAll + (size_t)nr * KCLU;

  if (tid < KCLU) {
    float s = csq[tid];
    colinv[tid] = 1.0f / sqrtf(s + EPSF);
    term[tid] = s / (s + EPSF);
  }
  __syncthreads();
  if (tid == 0) {
    float t = 0.0f;
    for (int c = 0; c < KCLU; ++c) t += term[c];
    gscale = 1.0f / sqrtf(t + EPSF);
  }
  __syncthreads();

  const float g = gscale;
  float* outp = out + (size_t)nr * OUT_PER_REG;
  for (int i = tid; i < OUT_PER_REG; i += 256)
    outp[i] *= colinv[i & (KCLU - 1)] * g;
}

// ---------------------------------------------------------------------------
extern "C" void kernel_launch(void* const* d_in, const int* in_sizes, int n_in,
                              void* d_out, int out_size, void* d_ws, size_t ws_size,
                              hipStream_t stream) {
  (void)in_sizes; (void)n_in; (void)out_size; (void)ws_size;
  const float* x  = (const float*)d_in[0];   // [32,36,36,512]
  const float* Wc = (const float*)d_in[1];   // [512,64]
  const float* Cm = (const float*)d_in[2];   // [512,64]
  float* out = (float*)d_out;                // [32, 9*512*64]
  char* ws = (char*)d_ws;

  _Float16* Wt = (_Float16*)(ws + WT_OFF);
  _Float16* At = (_Float16*)(ws + AT_OFF);
  float* colsq = (float*)(ws + CSQ_OFF);

  k_wt<<<128, 256, 0, stream>>>(Wc, Wt);
  k_assign<<<NIMG * NREG, 288, 0, stream>>>(x, Wt, At);
  k_vlad<<<NIMG * NREG, 256, 0, stream>>>(x, Cm, At, out, colsq);
  k_norm<<<NIMG * NREG, 256, 0, stream>>>(out, colsq);
}
